// Hamilton_V5_17076789969629
// MI455X (gfx1250) — matrix operand order repaired and checked by reading, not compiled, not run on hardware
//
#include <hip/hip_runtime.h>
#include <hip/hip_bf16.h>

typedef __attribute__((ext_vector_type(16))) _Float16 v16h;
typedef __attribute__((ext_vector_type(8)))  float    v8f;

#define WMMA_F16(a, b, c) \
  __builtin_amdgcn_wmma_f32_16x16x32_f16(false, (a), false, (b), (short)0, (c), false, false)

// ---------------------------------------------------------------------------
// Fragment loaders.
// A (MxK, row-major, 16-bit, 16x32 tile): lane l (lo=l&15, hi=l>>4) holds
//   row (mBase+lo), halves 0..15 = K (kBase + 16*hi + 0..15)
// B is stored TRANSPOSED (N x K "BT", row-major): lane l holds
//   col (nBase+lo), halves 0..15 = K (kBase + 16*hi + 0..15)
// Both are two 16-byte loads per lane (coalesced).
// ---------------------------------------------------------------------------
__device__ __forceinline__ v16h load_a_frag(const _Float16* __restrict__ A, int lda,
                                            int row, int kBase, int hi) {
  const _Float16* p = A + (size_t)row * lda + kBase + 16 * hi;
  v16h r;
  ((float4*)&r)[0] = *(const float4*)(p);
  ((float4*)&r)[1] = *(const float4*)(p + 8);
  return r;
}

__device__ __forceinline__ v16h load_b_frag(const _Float16* __restrict__ BT, int ldb,
                                            int col, int kBase, int hi) {
  const _Float16* p = BT + (size_t)col * ldb + kBase + 16 * hi;
  v16h r;
  ((float4*)&r)[0] = *(const float4*)(p);
  ((float4*)&r)[1] = *(const float4*)(p + 8);
  return r;
}

// ---------------------------------------------------------------------------
// Prep kernels
// ---------------------------------------------------------------------------
__global__ void convert_f16_kernel(const float* __restrict__ in,
                                   _Float16* __restrict__ out, int n) {
  int i = blockIdx.x * blockDim.x + threadIdx.x;
  if (i < n) out[i] = (_Float16)in[i];
}

// in: R x C (f32, row-major)  ->  out: C x R (f16, row-major). R,C multiples of 32.
__global__ void transpose_f16_kernel(const float* __restrict__ in,
                                     _Float16* __restrict__ out, int R, int C) {
  __shared__ _Float16 tile[32][33];
  const int c0 = blockIdx.x * 32, r0 = blockIdx.y * 32;
  const int tx = threadIdx.x, ty = threadIdx.y;  // (32, 8)
#pragma unroll
  for (int i = 0; i < 32; i += 8)
    tile[ty + i][tx] = (_Float16)in[(size_t)(r0 + ty + i) * C + (c0 + tx)];
  __syncthreads();
#pragma unroll
  for (int i = 0; i < 32; i += 8)
    out[(size_t)(c0 + ty + i) * R + (r0 + tx)] = tile[tx][ty + i];
}

// ---------------------------------------------------------------------------
// Generic GEMM + bias + tanh -> f16 output.
// Wave tile: 16(M) x 64(N). Block: 256 thr = 8 waves stacked along M (128 rows).
// Grid: (Ntotal/64, M/128). BT is N x K dense (ldb == K).
// ---------------------------------------------------------------------------
template <int K>
__global__ void __launch_bounds__(256)
gemm_tanh_kernel(const _Float16* __restrict__ A, int lda,
                 const _Float16* __restrict__ BT,
                 const float* __restrict__ bias,
                 _Float16* __restrict__ Out, int ldo) {
  const int tid  = threadIdx.x;
  const int wave = tid >> 5, l = tid & 31, lo = l & 15, hi = l >> 4;
  const int mBase = blockIdx.y * 128 + wave * 16;
  const int nBase = blockIdx.x * 64;

  v8f acc[4] = {};
  for (int k = 0; k < K; k += 32) {
    v16h a = load_a_frag(A, lda, mBase + lo, k, hi);
#pragma unroll
    for (int t = 0; t < 4; ++t) {
      v16h b = load_b_frag(BT, K, nBase + t * 16 + lo, k, hi);
      acc[t] = WMMA_F16(a, b, acc[t]);
    }
  }
#pragma unroll
  for (int t = 0; t < 4; ++t) {
    const int col = nBase + t * 16 + lo;
    const float bv = bias[col];
#pragma unroll
    for (int v = 0; v < 8; ++v) {
      const int m = mBase + 8 * hi + v;
      Out[(size_t)m * ldo + col] = (_Float16)tanhf(acc[t][v] + bv);
    }
  }
}

// ---------------------------------------------------------------------------
// Row reduction for H path: s[n] = sum_j h[n,j]*Hw2[j]; f[n] = 1 - tanh(s+Hb2)^2
// and in-place rewrite h -> w = Hw2 * (1 - h^2).
// ---------------------------------------------------------------------------
__global__ void hreduce_kernel(_Float16* __restrict__ h,
                               const float* __restrict__ Hw2,
                               const float* __restrict__ Hb2,
                               float* __restrict__ f) {
  const int n = blockIdx.x, tid = threadIdx.x;  // 128 threads
  float s = 0.f;
  for (int j = tid; j < 640; j += 128) {
    const float hv = (float)h[(size_t)n * 640 + j];
    const float w2 = Hw2[j];
    s += hv * w2;
    h[(size_t)n * 640 + j] = (_Float16)(w2 * (1.f - hv * hv));
  }
  __shared__ float red[128];
  red[tid] = s;
  __syncthreads();
  for (int off = 64; off > 0; off >>= 1) {
    if (tid < off) red[tid] += red[tid + off];
    __syncthreads();
  }
  if (tid == 0) {
    const float H = tanhf(red[0] + Hb2[0]);
    f[n] = 1.f - H * H;
  }
}

// ---------------------------------------------------------------------------
// G = f[m] * (W @ Hw1^T); out[m, col] = +g (col<128) / -g (col>=128), f32.
// BT here is Hw1 itself (256 x 640 f16, row-major) since (Hw1^T)^T = Hw1.
// ---------------------------------------------------------------------------
__global__ void __launch_bounds__(256)
gemm_g_kernel(const _Float16* __restrict__ W,       // 4096 x 640
              const _Float16* __restrict__ Hw1h,    // 256 x 640
              const float* __restrict__ f,
              float* __restrict__ Out) {
  constexpr int K = 640;
  const int tid  = threadIdx.x;
  const int wave = tid >> 5, l = tid & 31, lo = l & 15, hi = l >> 4;
  const int mBase = blockIdx.y * 128 + wave * 16;
  const int nBase = blockIdx.x * 64;

  v8f acc[4] = {};
  for (int k = 0; k < K; k += 32) {
    v16h a = load_a_frag(W, 640, mBase + lo, k, hi);
#pragma unroll
    for (int t = 0; t < 4; ++t) {
      v16h b = load_b_frag(Hw1h, K, nBase + t * 16 + lo, k, hi);
      acc[t] = WMMA_F16(a, b, acc[t]);
    }
  }
#pragma unroll
  for (int t = 0; t < 4; ++t) {
    const int col = nBase + t * 16 + lo;
    const float sgn = (col < 128) ? 1.f : -1.f;
#pragma unroll
    for (int v = 0; v < 8; ++v) {
      const int m = mBase + 8 * hi + v;
      Out[(size_t)m * 256 + col] = sgn * f[m] * acc[t][v];
    }
  }
}

// ---------------------------------------------------------------------------
// Fused big GEMM: C = h2 @ Aw3 (K=1024, Ntotal=16384), epilogue
//   dv[m, i] += sum_{j<128} tanh(C[m, i*128+j] + Ab3) * u[j]
//
// Block: 256 thr = 8 waves stacked along M; block tile 128(M) x 128(N).
// B tile (128 cols x 64 K-halves) is staged in LDS, double-buffered, shared by
// all 8 waves (8x cut in global/L2 B traffic). Rows padded to 72 halves so the
// 16-lane fragment reads (stride 36 dwords, gcd(36,64)=4) are bank-conflict
// free. One barrier per 64-deep k-block; next block's global loads overlap the
// current block's 16 WMMAs. All staging branches are wave-uniform (EXEC all-1s
// around WMMA).
// ---------------------------------------------------------------------------
#define A3_LDB 72  // padded LDS row stride in halves

__global__ void __launch_bounds__(256)
gemm_a3_kernel(const _Float16* __restrict__ h2,     // 4096 x 1024
               const _Float16* __restrict__ Aw3T,   // 16384 x 1024 (N-major)
               const float* __restrict__ Ab3,
               const float* __restrict__ u,
               float* __restrict__ Out) {
  __shared__ _Float16 Bs[2][128 * A3_LDB];  // 2 x 18432 B = 36 KB

  const int tid  = threadIdx.x;
  const int wave = tid >> 5, l = tid & 31, lo = l & 15, hi = l >> 4;
  const int mBase   = blockIdx.y * 128 + wave * 16;
  const int colBase = blockIdx.x * 128;

  // Cooperative staging: 128 rows x 64 halves = 1024 x 16B chunks, 4 per thread.
  // chunk c: row = c>>3, ci = c&7 (16B sub-chunk within the 128B row).
  float4 stg[4];
#pragma unroll
  for (int s = 0; s < 4; ++s) {
    const int c = tid + s * 256, row = c >> 3, ci = c & 7;
    stg[s] = *(const float4*)(Aw3T + (size_t)(colBase + row) * 1024 + ci * 8);
  }
#pragma unroll
  for (int s = 0; s < 4; ++s) {
    const int c = tid + s * 256, row = c >> 3, ci = c & 7;
    *(float4*)(&Bs[0][row * A3_LDB + ci * 8]) = stg[s];
  }
  __syncthreads();

  v8f acc[8] = {};
  int cur = 0;
  for (int kb = 0; kb < 1024; kb += 64) {
    const bool has_next = (kb + 64) < 1024;
    if (has_next) {  // issue next k-block's global loads early
#pragma unroll
      for (int s = 0; s < 4; ++s) {
        const int c = tid + s * 256, row = c >> 3, ci = c & 7;
        stg[s] = *(const float4*)(Aw3T + (size_t)(colBase + row) * 1024 + (kb + 64) + ci * 8);
      }
    }
#pragma unroll
    for (int ks = 0; ks < 64; ks += 32) {
      v16h a = load_a_frag(h2, 1024, mBase + lo, kb + ks, hi);
#pragma unroll
      for (int t = 0; t < 8; ++t) {
        const _Float16* p = &Bs[cur][(t * 16 + lo) * A3_LDB + ks + 16 * hi];
        v16h b;
        ((float4*)&b)[0] = *(const float4*)(p);
        ((float4*)&b)[1] = *(const float4*)(p + 8);
        acc[t] = WMMA_F16(a, b, acc[t]);
      }
    }
    if (has_next) {
#pragma unroll
      for (int s = 0; s < 4; ++s) {
        const int c = tid + s * 256, row = c >> 3, ci = c & 7;
        *(float4*)(&Bs[cur ^ 1][row * A3_LDB + ci * 8]) = stg[s];
      }
    }
    __syncthreads();
    cur ^= 1;
  }

  // Epilogue: tanh(acc + Ab3) * u, reduce over the 128 columns of this i-block.
  float p[8];
#pragma unroll
  for (int v = 0; v < 8; ++v) p[v] = 0.f;
#pragma unroll
  for (int t = 0; t < 8; ++t) {
    const int col = colBase + t * 16 + lo;
    const float bv = Ab3[col];
    const float uv = u[t * 16 + lo];
#pragma unroll
    for (int v = 0; v < 8; ++v) p[v] += tanhf(acc[t][v] + bv) * uv;
  }
#pragma unroll
  for (int v = 0; v < 8; ++v) {
    float s = p[v];
    s += __shfl_xor(s, 1);
    s += __shfl_xor(s, 2);
    s += __shfl_xor(s, 4);
    s += __shfl_xor(s, 8);
    p[v] = s;
  }
  if (lo == 0) {  // lane 0 -> rows 0..7, lane 16 -> rows 8..15
    const int i = blockIdx.x;
#pragma unroll
    for (int v = 0; v < 8; ++v) {
      const int m = mBase + 8 * hi + v;
      Out[(size_t)m * 256 + 128 + i] += p[v];  // unique writer per (m,i)
    }
  }
}

// ---------------------------------------------------------------------------
// Host launcher
// ---------------------------------------------------------------------------
extern "C" void kernel_launch(void* const* d_in, const int* in_sizes, int n_in,
                              void* d_out, int out_size, void* d_ws, size_t ws_size,
                              hipStream_t stream) {
  // setup_inputs order: t, inp, Hw1, Hb1, Hw2, Hb2, Aw1, Ab1, Aw2, Ab2, Aw3, Ab3, u
  const float* inp = (const float*)d_in[1];   // 4096 x 256
  const float* Hw1 = (const float*)d_in[2];   // 256 x 640
  const float* Hb1 = (const float*)d_in[3];   // 640
  const float* Hw2 = (const float*)d_in[4];   // 640 x 1
  const float* Hb2 = (const float*)d_in[5];   // 1
  const float* Aw1 = (const float*)d_in[6];   // 128 x 512
  const float* Ab1 = (const float*)d_in[7];   // 512
  const float* Aw2 = (const float*)d_in[8];   // 512 x 1024
  const float* Ab2 = (const float*)d_in[9];   // 1024
  const float* Aw3 = (const float*)d_in[10];  // 1024 x 16384
  const float* Ab3 = (const float*)d_in[11];  // 16384
  const float* u   = (const float*)d_in[12];  // 128
  float* out = (float*)d_out;                 // 4096 x 256

  char* ws = (char*)d_ws;
  _Float16* inp16 = (_Float16*)(ws);              // 4096*256*2  = 2,097,152
  _Float16* Hw1h  = (_Float16*)(ws + 2097152);    // 256*640*2   =   327,680
  _Float16* Hw1T  = (_Float16*)(ws + 2424832);    // 640*256*2   =   327,680
  _Float16* Aw1T  = (_Float16*)(ws + 2752512);    // 512*128*2   =   131,072
  _Float16* Aw2T  = (_Float16*)(ws + 2883584);    // 1024*512*2  = 1,048,576
  _Float16* Aw3T  = (_Float16*)(ws + 3932160);    // 16384*1024*2= 33,554,432
  _Float16* hbuf  = (_Float16*)(ws + 37486592);   // 4096*640*2  = 5,242,880
  _Float16* h1    = (_Float16*)(ws + 42729472);   // 4096*512*2  = 4,194,304
  _Float16* h2    = (_Float16*)(ws + 46923776);   // 4096*1024*2 = 8,388,608
  float*    fbuf  = (float*)(ws + 55312384);      // 4096*4      =    16,384
  (void)in_sizes; (void)n_in; (void)out_size; (void)ws_size;

  // 1) precision conversion + weight transposes (B stored N-major for WMMA)
  convert_f16_kernel<<<(1048576 + 255) / 256, 256, 0, stream>>>(inp, inp16, 1048576);
  convert_f16_kernel<<<(163840 + 255) / 256, 256, 0, stream>>>(Hw1, Hw1h, 163840);
  transpose_f16_kernel<<<dim3(640 / 32, 256 / 32), dim3(32, 8), 0, stream>>>(Hw1, Hw1T, 256, 640);
  transpose_f16_kernel<<<dim3(512 / 32, 128 / 32), dim3(32, 8), 0, stream>>>(Aw1, Aw1T, 128, 512);
  transpose_f16_kernel<<<dim3(1024 / 32, 512 / 32), dim3(32, 8), 0, stream>>>(Aw2, Aw2T, 512, 1024);
  transpose_f16_kernel<<<dim3(16384 / 32, 1024 / 32), dim3(32, 8), 0, stream>>>(Aw3, Aw3T, 1024, 16384);

  // 2) H path: h = tanh(z@Hw1+Hb1); s-reduction; g = f * (w @ Hw1^T) -> out (+/-)
  gemm_tanh_kernel<256><<<dim3(10, 32), 256, 0, stream>>>(inp16, 256, Hw1T, Hb1, hbuf, 640);
  hreduce_kernel<<<4096, 128, 0, stream>>>(hbuf, Hw2, Hb2, fbuf);
  gemm_g_kernel<<<dim3(4, 32), 256, 0, stream>>>(hbuf, Hw1h, fbuf, out);

  // 3) A path: h1, h2, then fused big GEMM with tanh * u reduction into out[:,128:]
  gemm_tanh_kernel<128><<<dim3(8, 32), 256, 0, stream>>>(inp16, 256, Aw1T, Ab1, h1, 512);
  gemm_tanh_kernel<512><<<dim3(16, 32), 256, 0, stream>>>(h1, 512, Aw2T, Ab2, h2, 1024);
  gemm_a3_kernel<<<dim3(128, 32), 256, 0, stream>>>(h2, Aw3T, Ab3, u, out);
}
